// SD_attn_16131897164209
// MI455X (gfx1250) — matrix operand order repaired and checked by reading, not compiled, not run on hardware
//
#include <hip/hip_runtime.h>
#include <hip/hip_bf16.h>

typedef __bf16 bf16;
typedef __attribute__((ext_vector_type(16))) __bf16 v16bf;
typedef __attribute__((ext_vector_type(8)))  float  v8f;
typedef __attribute__((ext_vector_type(4)))  unsigned int u32x4;
typedef __attribute__((ext_vector_type(2)))  unsigned int u32x2;

union AB16 {
    v16bf v;
    unsigned int w[8];
    u32x4 q[2];
};

__device__ __forceinline__ unsigned int pk_bf16(float a, float b) {
    union { __bf16 h[2]; unsigned int u; } t;
    t.h[0] = (bf16)a;
    t.h[1] = (bf16)b;
    return t.u;
}

// ---------------------------------------------------------------------------
// WMMA helpers (CDNA5 16x16x32 bf16, f32 accumulate)
// ---------------------------------------------------------------------------
__device__ __forceinline__ v8f wmma_bf16(v16bf a, v16bf b, v8f c) {
    return __builtin_amdgcn_wmma_f32_16x16x32_bf16(
        /*neg_a=*/false, a, /*neg_b=*/false, b,
        /*c_mod=*/(short)0, c, /*reuse_a=*/false, /*reuse_b=*/false);
}

// A-matrix (16x32 bf16) load from row-major LDS tile; contiguous-K layout (same k order as the B loads):
// lane = g*16 + m ; VGPR v holds K pair k0=g*16 + v*2
__device__ __forceinline__ v16bf load_A_lds(const bf16* lds, int m0, int ld,
                                            int kbase, int lane) {
    int m = m0 + (lane & 15);
    int g = lane >> 4;
    AB16 a;
#pragma unroll
    for (int vv = 0; vv < 8; ++vv) {
        int k0 = (g << 4) + (vv << 1);
        a.w[vv] = *(const unsigned int*)(lds + m * ld + kbase + k0);
    }
    return a.v;
}

// B-matrix (32x16 bf16) load: lane = g*16 + n holds K = kbase + g*16 + [0..15]
// contiguous -> works directly on row-major W[n][k] with row stride ld.
__device__ __forceinline__ v16bf load_B_glb(const bf16* W, int n0, int ld,
                                            int kbase, int lane) {
    int n = n0 + (lane & 15);
    int g = lane >> 4;
    const bf16* p = W + (size_t)n * ld + kbase + (g << 4);
    AB16 b;
    b.q[0] = *(const u32x4*)(p);
    b.q[1] = *(const u32x4*)(p + 8);
    return b.v;
}

__device__ __forceinline__ v16bf load_B_lds(const bf16* lds, int n0, int ld,
                                            int kbase, int lane) {
    int n = n0 + (lane & 15);
    int g = lane >> 4;
    const bf16* p = lds + n * ld + kbase + (g << 4);
    AB16 b;
    b.q[0] = *(const u32x4*)(p);
    b.q[1] = *(const u32x4*)(p + 8);
    return b.v;
}

// ---------------------------------------------------------------------------
// Kernel 0: convert weights fp32 -> bf16 (vectorized, 4 elements/thread)
// ---------------------------------------------------------------------------
__global__ __launch_bounds__(256) void cvt_kernel(const float* __restrict__ qkvw,
                                                  const float* __restrict__ projw,
                                                  bf16* qkvwb, bf16* projwb) {
    int i4 = (blockIdx.x * 256 + threadIdx.x) * 4;
    if (i4 < 1536 * 512) {
        float4 f = *(const float4*)(qkvw + i4);
        u32x2 pk;
        pk.x = pk_bf16(f.x, f.y);
        pk.y = pk_bf16(f.z, f.w);
        *(u32x2*)(qkvwb + i4) = pk;
    }
    if (i4 < 512 * 512) {
        float4 f = *(const float4*)(projw + i4);
        u32x2 pk;
        pk.x = pk_bf16(f.x, f.y);
        pk.y = pk_bf16(f.z, f.w);
        *(u32x2*)(projwb + i4) = pk;
    }
}

// ---------------------------------------------------------------------------
// Kernel 1: shift + window-partition + QKV GEMM (64x512 @ 512x1536) per window
// 256 threads = 8 waves, one window per block. LDS holds x tile (bf16, 64KB).
// ---------------------------------------------------------------------------
__global__ __launch_bounds__(256) void qkv_kernel(const float* __restrict__ x,
                                                  const bf16* __restrict__ Wb,
                                                  const float* __restrict__ bias,
                                                  bf16* qws, bf16* kws, bf16* vws) {
    extern __shared__ bf16 xs[];  // 64*512 bf16
    int win = blockIdx.x;
    int b   = win >> 8;
    int wh  = (win >> 4) & 15;
    int ww  = win & 15;
    int tid = threadIdx.x;

    // Stage shifted window into LDS (roll by -4,-4 => src = (dst+4) mod 128).
    // float4 loads, packed-bf16 8B LDS stores.
    for (int idx = tid; idx < 64 * 128; idx += 256) {
        int t  = idx >> 7;
        int d4 = (idx & 127) << 2;
        int r = t >> 3, c = t & 7;
        int h = (wh * 8 + r + 4) & 127;
        int w = (ww * 8 + c + 4) & 127;
        float4 f = *(const float4*)(x + (((size_t)b * 128 + h) * 128 + w) * 512 + d4);
        u32x2 pk;
        pk.x = pk_bf16(f.x, f.y);
        pk.y = pk_bf16(f.z, f.w);
        *(u32x2*)(xs + t * 512 + d4) = pk;
    }

    int wave = tid >> 5, lane = tid & 31;
    // Warm L2 with the weight rows this wave touches first (global_prefetch_b8)
    __builtin_prefetch(Wb + (size_t)(wave * 16 + (lane & 15)) * 512, 0, 1);
    __syncthreads();

    for (int qq = 0; qq < 12; ++qq) {
        int nt = wave + (qq << 3);  // 0..95 column tiles of 1536
        int n0 = nt << 4;
        for (int mt = 0; mt < 4; ++mt) {
            v8f acc = {0.f, 0.f, 0.f, 0.f, 0.f, 0.f, 0.f, 0.f};
#pragma unroll 4
            for (int kt = 0; kt < 16; ++kt) {
                v16bf a  = load_A_lds(xs, mt * 16, 512, kt * 32, lane);
                v16bf bm = load_B_glb(Wb, n0, 512, kt * 32, lane);
                acc = wmma_bf16(a, bm, acc);
            }
            int   j0 = n0 + (lane & 15);  // output column 0..1535
            float bj = bias[j0];
            bf16* buf = (j0 < 512) ? qws : (j0 < 1024 ? kws : vws);
            int   jj = j0 & 511;
            int   head = jj >> 6, d = jj & 63;
#pragma unroll
            for (int r = 0; r < 8; ++r) {
                int i = mt * 16 + r + ((lane >> 4) << 3);
                buf[(((size_t)win * 8 + head) * 64 + i) * 64 + d] =
                    (bf16)(acc[r] + bj);
            }
        }
    }
}

// ---------------------------------------------------------------------------
// Kernel 2: per (window, head) attention. 128 threads = 4 waves.
//  - stage q,k with RoPE (q also * SCALE), v transposed, into LDS
//  - S = q k^T (WMMA), +mask (wh==15 only), row softmax in registers,
//  - P -> LDS -> A layout, O = P V (WMMA), store bf16 [win][tok][head*64+d]
// RoPE couples dims (d, d+16) with the same angle -> one sincos per pair.
// ---------------------------------------------------------------------------
__global__ __launch_bounds__(128) void attn_kernel(const bf16* __restrict__ qws,
                                                   const bf16* __restrict__ kws,
                                                   const bf16* __restrict__ vws,
                                                   bf16* __restrict__ aws) {
    __shared__ bf16 qs[64 * 64];
    __shared__ bf16 ks[64 * 64];
    __shared__ bf16 vT[64 * 64];
    __shared__ bf16 ps[64 * 64];
    __shared__ float freqs[16];

    int blk  = blockIdx.x;
    int win  = blk >> 3;
    int head = blk & 7;
    int wh   = (win >> 4) & 15;
    size_t base = ((size_t)win * 8 + head) * 4096;
    const bf16* qsrc = qws + base;
    const bf16* ksrc = kws + base;
    const bf16* vsrc = vws + base;

    int tid = threadIdx.x;
    if (tid < 16) freqs[tid] = __powf(10000.f, -(float)tid * (1.f / 16.f));
    __syncthreads();

    // q/k RoPE: 2048 (token, pair) units; pair = (d, d+16) within each half.
    for (int idx = tid; idx < 2048; idx += 128) {
        int t = idx >> 5;
        int p = idx & 31;
        int d  = (p < 16) ? p : (p + 16);  // low half of h-part / w-part
        int d2 = d + 16;
        float pos = (d < 32) ? (float)(t >> 3) : (float)(t & 7);
        float ang = pos * freqs[d & 15];
        float s, c;
        __sincosf(ang, &s, &c);
        float qa = (float)qsrc[t * 64 + d], qb = (float)qsrc[t * 64 + d2];
        float ka = (float)ksrc[t * 64 + d], kb = (float)ksrc[t * 64 + d2];
        qs[t * 64 + d ] = (bf16)((qa * c - qb * s) * 0.125f);  // SCALE=64^-0.5
        qs[t * 64 + d2] = (bf16)((qb * c + qa * s) * 0.125f);
        ks[t * 64 + d ] = (bf16)(ka * c - kb * s);
        ks[t * 64 + d2] = (bf16)(kb * c + ka * s);
    }
    // v transposed so the B operand reads contiguous-K
    for (int idx = tid; idx < 4096; idx += 128) {
        int t = idx >> 6, d = idx & 63;
        vT[d * 64 + t] = vsrc[idx];
    }
    __syncthreads();

    int wave = tid >> 5, lane = tid & 31;
    int mt = wave;  // each wave owns a 16-row strip of the 64x64 score matrix

    v8f acc[4];
#pragma unroll
    for (int nt = 0; nt < 4; ++nt)
        acc[nt] = (v8f){0.f, 0.f, 0.f, 0.f, 0.f, 0.f, 0.f, 0.f};

#pragma unroll
    for (int nt = 0; nt < 4; ++nt)
#pragma unroll
        for (int kt = 0; kt < 2; ++kt)
            acc[nt] = wmma_bf16(load_A_lds(qs, mt * 16, 64, kt * 32, lane),
                                load_B_lds(ks, nt * 16, 64, kt * 32, lane),
                                acc[nt]);

    // Shift mask: reference's w_slices make label depend only on shifted row;
    // only the last window row (wh==15) mixes regions (token row <4 vs >=4).
    if (wh == 15) {
#pragma unroll
        for (int nt = 0; nt < 4; ++nt)
#pragma unroll
            for (int r = 0; r < 8; ++r) {
                int i = mt * 16 + r + ((lane >> 4) << 3);
                int j = nt * 16 + (lane & 15);
                if (((i >> 3) < 4) != ((j >> 3) < 4)) acc[nt][r] += -100.f;
            }
    }

    // Row softmax: each (r, lane-group) owns one row; cols live across the
    // 16 lanes of the group over the 4 nt tiles. xor masks 1,2,4,8 stay
    // within a 16-lane group.
#pragma unroll
    for (int r = 0; r < 8; ++r) {
        float mx = fmaxf(fmaxf(acc[0][r], acc[1][r]), fmaxf(acc[2][r], acc[3][r]));
#pragma unroll
        for (int m = 1; m < 16; m <<= 1) mx = fmaxf(mx, __shfl_xor(mx, m, 32));
        float sum = 0.f;
#pragma unroll
        for (int nt = 0; nt < 4; ++nt) {
            acc[nt][r] = __expf(acc[nt][r] - mx);
            sum += acc[nt][r];
        }
#pragma unroll
        for (int m = 1; m < 16; m <<= 1) sum += __shfl_xor(sum, m, 32);
        float inv = 1.f / sum;
#pragma unroll
        for (int nt = 0; nt < 4; ++nt) acc[nt][r] *= inv;
    }

    // P -> LDS (row major) so it can be re-read in A layout
#pragma unroll
    for (int nt = 0; nt < 4; ++nt)
#pragma unroll
        for (int r = 0; r < 8; ++r) {
            int i = mt * 16 + r + ((lane >> 4) << 3);
            int j = nt * 16 + (lane & 15);
            ps[i * 64 + j] = (bf16)acc[nt][r];
        }
    __syncthreads();

    // O = P @ V
    v8f oacc[4];
#pragma unroll
    for (int nt = 0; nt < 4; ++nt)
        oacc[nt] = (v8f){0.f, 0.f, 0.f, 0.f, 0.f, 0.f, 0.f, 0.f};
#pragma unroll
    for (int nt = 0; nt < 4; ++nt)
#pragma unroll
        for (int kt = 0; kt < 2; ++kt)
            oacc[nt] = wmma_bf16(load_A_lds(ps, mt * 16, 64, kt * 32, lane),
                                 load_B_lds(vT, nt * 16, 64, kt * 32, lane),
                                 oacc[nt]);

    bf16* dst = aws + (size_t)win * 64 * 512 + head * 64;
#pragma unroll
    for (int nt = 0; nt < 4; ++nt)
#pragma unroll
        for (int r = 0; r < 8; ++r) {
            int i = mt * 16 + r + ((lane >> 4) << 3);
            int j = nt * 16 + (lane & 15);
            dst[(size_t)i * 512 + j] = (bf16)oacc[nt][r];
        }
}

// ---------------------------------------------------------------------------
// Kernel 3: projection GEMM (64x512 @ 512x512) + bias + unpartition + unroll
// ---------------------------------------------------------------------------
__global__ __launch_bounds__(256) void proj_kernel(const bf16* __restrict__ aws,
                                                   const bf16* __restrict__ Wb,
                                                   const float* __restrict__ bias,
                                                   float* __restrict__ out) {
    extern __shared__ bf16 as_[];  // 64*512 bf16
    int win = blockIdx.x;
    int b   = win >> 8;
    int wh  = (win >> 4) & 15;
    int ww  = win & 15;
    int tid = threadIdx.x;

    // 16-byte vector copy of the attention tile into LDS
    {
        const u32x4* s4 = (const u32x4*)(aws + (size_t)win * 64 * 512);
        u32x4* d4 = (u32x4*)as_;
        for (int i = tid; i < 4096; i += 256) d4[i] = s4[i];
    }

    int wave = tid >> 5, lane = tid & 31;
    __builtin_prefetch(Wb + (size_t)(wave * 16 + (lane & 15)) * 512, 0, 1);
    __syncthreads();

    for (int qq = 0; qq < 4; ++qq) {
        int nt = wave + (qq << 3);  // 0..31
        int n0 = nt << 4;
        for (int mt = 0; mt < 4; ++mt) {
            v8f acc = {0.f, 0.f, 0.f, 0.f, 0.f, 0.f, 0.f, 0.f};
#pragma unroll 4
            for (int kt = 0; kt < 16; ++kt) {
                v16bf a  = load_A_lds(as_, mt * 16, 512, kt * 32, lane);
                v16bf bm = load_B_glb(Wb, n0, 512, kt * 32, lane);
                acc = wmma_bf16(a, bm, acc);
            }
            int   j  = n0 + (lane & 15);
            float bj = bias[j];
#pragma unroll
            for (int r = 0; r < 8; ++r) {
                int i  = mt * 16 + r + ((lane >> 4) << 3);
                int rr = i >> 3, cc = i & 7;
                int h  = (wh * 8 + rr + 4) & 127;  // roll back by +4
                int w  = (ww * 8 + cc + 4) & 127;
                out[(((size_t)b * 128 + h) * 128 + w) * 512 + j] = acc[r] + bj;
            }
        }
    }
}

// ---------------------------------------------------------------------------
// Launch
// ---------------------------------------------------------------------------
extern "C" void kernel_launch(void* const* d_in, const int* in_sizes, int n_in,
                              void* d_out, int out_size, void* d_ws, size_t ws_size,
                              hipStream_t stream) {
    const float* x      = (const float*)d_in[0];
    const float* qkv_w  = (const float*)d_in[1];
    const float* qkv_b  = (const float*)d_in[2];
    const float* proj_w = (const float*)d_in[3];
    const float* proj_b = (const float*)d_in[4];
    float* out = (float*)d_out;

    char* ws = (char*)d_ws;
    const size_t QKVW_B  = (size_t)1536 * 512 * 2;          // 1.5 MB
    const size_t HEADBUF = (size_t)2048 * 8 * 64 * 64 * 2;  // 128 MB each
    bf16* qkvwb  = (bf16*)(ws);
    bf16* projwb = (bf16*)(ws + QKVW_B);
    bf16* qws = (bf16*)(ws + 2097152);
    bf16* kws = (bf16*)(ws + 2097152 + HEADBUF);
    bf16* vws = (bf16*)(ws + 2097152 + 2 * HEADBUF);
    bf16* aws = (bf16*)(ws + 2097152 + 3 * HEADBUF);
    (void)in_sizes; (void)n_in; (void)out_size; (void)ws_size;

    cvt_kernel<<<768, 256, 0, stream>>>(qkv_w, proj_w, qkvwb, projwb);
    qkv_kernel<<<2048, 256, 64 * 1024, stream>>>(x, qkvwb, qkv_b, qws, kws, vws);
    attn_kernel<<<2048 * 8, 128, 0, stream>>>(qws, kws, vws, aws);
    proj_kernel<<<2048, 256, 64 * 1024, stream>>>(aws, projwb, proj_b, out);
}
